// LaneGraphConvLayerShared_55722905698616
// MI455X (gfx1250) — matrix operand order repaired and checked by reading, not compiled, not run on hardware
//
#include <hip/hip_runtime.h>

typedef __attribute__((ext_vector_type(16))) _Float16 v16h;
typedef __attribute__((ext_vector_type(8)))  float    v8f;

#define NN    50000
#define EE    150000
#define DD    64
#define EDIM  16
#define NCOLS 1152   // 16*64 (edge basis) + 64 (bias matrix) + 64 (W_root + W_res)

// ---------------------------------------------------------------------------
// Kernel 1: rearrange weights into a column-major (K-contiguous) f16 matrix
// WcatT[n][k], n = output column of the big GEMM, k = contraction index d.
//   n <  1024 : W_edge[k_edge][d*64+f] with n = k_edge*64 + f
//   n <  1088 : bias matrix  b_edge[d*64+f],        f = n - 1024
//   else      : (W_root + W_res)[d][f],             f = n - 1088
// Also b_comb = bias + b_res.
// ---------------------------------------------------------------------------
__global__ void __launch_bounds__(256)
prep_weights(const float* __restrict__ W_edge, const float* __restrict__ b_edge,
             const float* __restrict__ W_root, const float* __restrict__ bias,
             const float* __restrict__ W_res,  const float* __restrict__ b_res,
             _Float16* __restrict__ WcatT, float* __restrict__ bcomb) {
  int idx = blockIdx.x * blockDim.x + threadIdx.x;
  if (idx < NCOLS * DD) {
    int n = idx >> 6;
    int k = idx & 63;
    float v;
    if (n < 1024) {
      int kk = n >> 6, f = n & 63;
      v = W_edge[kk * 4096 + k * 64 + f];
    } else if (n < 1088) {
      int f = n - 1024;
      v = b_edge[k * 64 + f];
    } else {
      int f = n - 1088;
      v = W_root[k * 64 + f] + W_res[k * 64 + f];
    }
    WcatT[n * 64 + k] = (_Float16)v;
  }
  if (idx < DD) bcomb[idx] = bias[idx] + b_res[idx];
}

// ---------------------------------------------------------------------------
// Kernel 2: zero the scatter-add accumulator (atomics accumulate into it).
// ---------------------------------------------------------------------------
__global__ void __launch_bounds__(256)
zero_agg(float* __restrict__ agg) {
  int idx = blockIdx.x * blockDim.x + threadIdx.x;
  if (idx < NN * DD) agg[idx] = 0.f;
}

// ---------------------------------------------------------------------------
// Kernel 3: WMMA GEMM  Xall[N,1152] = f16( x[N,64] @ Wcat[64,1152] )
// 16x16x32 f16 WMMA, K=64 -> 2 WMMAs per tile. One block = 16 rows, 4 waves,
// each wave sweeps 72/4 = 18 column tiles reusing its A fragment.
// A fragment (16-bit A 16x32): lane<16 -> row=lane, K={kb..kb+15}
// with kb=0; lane>=16 -> same rows, kb=16.  B fragment: lane column = lane&15,
// 16 contiguous K halves starting at (lane<16 ? 0 : 16) -- WcatT is K-major so
// this is one aligned 32-byte load.  D: VGPR j -> row j + 8*(lane>=16).
// ---------------------------------------------------------------------------
__global__ void __launch_bounds__(128)
node_gemm(const float* __restrict__ x, const _Float16* __restrict__ WcatT,
          _Float16* __restrict__ Xall) {
  const int m0   = blockIdx.x * 16;
  const int wave = threadIdx.x >> 5;
  const int lane = threadIdx.x & 31;
  const int mrow = lane & 15;
  const bool hi  = lane >= 16;
  const int kb   = hi ? 16 : 0;

  const float* xr = x + (size_t)(m0 + mrow) * DD;
  v16h a0, a1;
#pragma unroll
  for (int i = 0; i < 8; ++i) {
    a0[i]     = (_Float16)xr[kb + i];
    a0[i + 8] = (_Float16)xr[kb + 8 + i];
    a1[i]     = (_Float16)xr[32 + kb + i];
    a1[i + 8] = (_Float16)xr[32 + kb + 8 + i];
  }

  const int kg = hi ? 16 : 0;
  for (int nt = wave; nt < NCOLS / 16; nt += 4) {
    const int c = nt * 16 + mrow;
    const _Float16* wc = WcatT + c * 64 + kg;
    v16h b0 = *(const v16h*)(wc);        // K  0..31 slice for this lane group
    v16h b1 = *(const v16h*)(wc + 32);   // K 32..63 slice
    v8f acc = {};
    acc = __builtin_amdgcn_wmma_f32_16x16x32_f16(false, a0, false, b0,
                                                 (short)0, acc, false, false);
    acc = __builtin_amdgcn_wmma_f32_16x16x32_f16(false, a1, false, b1,
                                                 (short)0, acc, false, false);
    _Float16* outp = Xall + (size_t)(m0 + (hi ? 8 : 0)) * NCOLS + c;
#pragma unroll
    for (int j = 0; j < 8; ++j)
      outp[(size_t)j * NCOLS] = (_Float16)acc[j];
  }
}

// ---------------------------------------------------------------------------
// Kernel 4: per-edge message + scatter-add.  One wave per edge, 2 features
// per lane.  msg = Sigma_k a_k * Y_k[src] + Y_bias[src]; atomic add to dst.
// ---------------------------------------------------------------------------
__global__ void __launch_bounds__(256)
edge_msg(const int* __restrict__ edge_index, const float* __restrict__ edge_attr,
         const _Float16* __restrict__ Xall, float* __restrict__ agg) {
  const int wid = blockIdx.x * (blockDim.x >> 5) + (threadIdx.x >> 5);
  if (wid >= EE) return;
  const int lane = threadIdx.x & 31;
  const int src = edge_index[wid];
  const int dst = edge_index[EE + wid];
  const _Float16* Y = Xall + (size_t)src * NCOLS;
  const int f0 = lane * 2;

  float av = edge_attr[wid * EDIM + (lane & 15)];
  float acc0 = (float)Y[1024 + f0];       // bias-matrix contribution
  float acc1 = (float)Y[1024 + f0 + 1];
#pragma unroll
  for (int k = 0; k < EDIM; ++k) {
    float a = __shfl(av, k, 32);
    acc0 += a * (float)Y[k * 64 + f0];
    acc1 += a * (float)Y[k * 64 + f0 + 1];
  }
  atomicAdd(&agg[(size_t)dst * DD + f0],     acc0);
  atomicAdd(&agg[(size_t)dst * DD + f0 + 1], acc1);
}

// ---------------------------------------------------------------------------
// Kernel 5: h = agg + x@(W_root+W_res) (columns 1088..1151 of Xall) + b_comb,
// then LayerNorm over D=64.  One wave per node, wave32 shuffle reductions.
// ---------------------------------------------------------------------------
__global__ void __launch_bounds__(256)
node_finish(const float* __restrict__ agg, const _Float16* __restrict__ Xall,
            const float* __restrict__ bcomb, const float* __restrict__ gamma,
            const float* __restrict__ beta, float* __restrict__ out) {
  const int n = blockIdx.x * (blockDim.x >> 5) + (threadIdx.x >> 5);
  if (n >= NN) return;
  const int lane = threadIdx.x & 31;
  const int f0 = lane * 2, f1 = f0 + 1;

  float h0 = agg[(size_t)n * DD + f0] +
             (float)Xall[(size_t)n * NCOLS + 1088 + f0] + bcomb[f0];
  float h1 = agg[(size_t)n * DD + f1] +
             (float)Xall[(size_t)n * NCOLS + 1088 + f1] + bcomb[f1];

  float s = h0 + h1;
#pragma unroll
  for (int off = 16; off >= 1; off >>= 1) s += __shfl_xor(s, off, 32);
  const float mu = s * (1.f / 64.f);

  float d0 = h0 - mu, d1 = h1 - mu;
  float v = d0 * d0 + d1 * d1;
#pragma unroll
  for (int off = 16; off >= 1; off >>= 1) v += __shfl_xor(v, off, 32);
  const float inv = rsqrtf(v * (1.f / 64.f) + 1e-5f);

  out[(size_t)n * DD + f0] = d0 * inv * gamma[f0] + beta[f0];
  out[(size_t)n * DD + f1] = d1 * inv * gamma[f1] + beta[f1];
}

// ---------------------------------------------------------------------------
// Workspace layout (bytes):
//   [0,        147456)   WcatT  f16 [1152][64]
//   [147456,   147712)   bcomb  f32 [64]
//   [147712, 12947712)   agg    f32 [50000][64]
//   [12947712,128147712) Xall   f16 [50000][1152]
// ---------------------------------------------------------------------------
extern "C" void kernel_launch(void* const* d_in, const int* in_sizes, int n_in,
                              void* d_out, int out_size, void* d_ws, size_t ws_size,
                              hipStream_t stream) {
  const float* x          = (const float*)d_in[0];
  const int*   edge_index = (const int*)  d_in[1];
  const float* edge_attr  = (const float*)d_in[2];
  const float* W_edge     = (const float*)d_in[3];
  const float* b_edge     = (const float*)d_in[4];
  const float* W_root     = (const float*)d_in[5];
  const float* bias       = (const float*)d_in[6];
  const float* W_res      = (const float*)d_in[7];
  const float* b_res      = (const float*)d_in[8];
  const float* gamma      = (const float*)d_in[9];
  const float* beta       = (const float*)d_in[10];
  float* out = (float*)d_out;

  char* ws = (char*)d_ws;
  _Float16* WcatT = (_Float16*)(ws);
  float*    bcomb = (float*)   (ws + 147456);
  float*    agg   = (float*)   (ws + 147712);
  _Float16* Xall  = (_Float16*)(ws + 12947712);

  prep_weights<<<(NCOLS * DD + 255) / 256, 256, 0, stream>>>(
      W_edge, b_edge, W_root, bias, W_res, b_res, WcatT, bcomb);
  zero_agg<<<(NN * DD + 255) / 256, 256, 0, stream>>>(agg);
  node_gemm<<<NN / 16, 128, 0, stream>>>(x, WcatT, Xall);
  edge_msg<<<(EE + 7) / 8, 256, 0, stream>>>(edge_index, edge_attr, Xall, agg);
  node_finish<<<(NN + 7) / 8, 256, 0, stream>>>(agg, Xall, bcomb, gamma, beta, out);
}
